// MultiHeadAttention_11278584119523
// MI455X (gfx1250) — matrix operand order repaired and checked by reading, not compiled, not run on hardware
//
#include <hip/hip_runtime.h>

// CDNA5 / gfx1250: wave32, WMMA 16x16x32 bf16 -> f32 accumulate.

typedef __attribute__((ext_vector_type(16))) __bf16 v16bf;
typedef __attribute__((ext_vector_type(8)))  float  v8f;

union Frag { v16bf v; uint4 q[2]; };
union BfPack8 { __bf16 h[8]; uint4 q; };

__device__ __forceinline__ __bf16 f2bf(float f) {
  unsigned u = __builtin_bit_cast(unsigned, f);
  u += 0x7fffu + ((u >> 16) & 1u);          // round-to-nearest-even
  unsigned short h = (unsigned short)(u >> 16);
  return __builtin_bit_cast(__bf16, h);
}

// ---------------------------------------------------------------- converts
// 8 fp32 -> 8 bf16 per thread: 2x b128 loads, 1x b128 store.
__global__ void cvt_bf16_kernel(const float* __restrict__ in,
                                __bf16* __restrict__ out, int n) {
  int i = (blockIdx.x * blockDim.x + threadIdx.x) * 8;
  if (i >= n) return;
  float4 a = *(const float4*)(in + i);
  float4 b = *(const float4*)(in + i + 4);
  BfPack8 p;
  p.h[0] = f2bf(a.x); p.h[1] = f2bf(a.y); p.h[2] = f2bf(a.z); p.h[3] = f2bf(a.w);
  p.h[4] = f2bf(b.x); p.h[5] = f2bf(b.y); p.h[6] = f2bf(b.z); p.h[7] = f2bf(b.w);
  *(uint4*)(out + i) = p.q;
}

// in [K][N] fp32 row-major -> out [N][K] bf16; 32x32 LDS tile, coalesced
// on both the read and the write side.
__global__ __launch_bounds__(256) void transpose_cvt_kernel(
    const float* __restrict__ in, __bf16* __restrict__ out, int K, int N) {
  __shared__ float tile[32][33];
  const int n0 = blockIdx.x * 32, k0 = blockIdx.y * 32;
  const int tx = threadIdx.x, ty = threadIdx.y;
#pragma unroll
  for (int i = ty; i < 32; i += 8)
    tile[i][tx] = in[(size_t)(k0 + i) * N + n0 + tx];
  __syncthreads();
#pragma unroll
  for (int i = ty; i < 32; i += 8)
    out[(size_t)(n0 + i) * K + k0 + tx] = f2bf(tile[tx][i]);
}

// ------------------------------------------------- shared GEMM building blocks
// Wave-level compute step: 2x4 WMMA tiles from one staged 128x32 K-slice.
__device__ __forceinline__ void gemm_compute_step(
    const __bf16* __restrict__ Ab, const __bf16* __restrict__ Bb,
    int wm, int wn, int lm, int hi, v8f acc[2][4]) {
  Frag a[2], b[4];
#pragma unroll
  for (int mt = 0; mt < 2; mt++) {
    // A 16x32 layout: lane m=lm; chunks K[klo..klo+8), K[16+klo..+8), klo=hi*8
    const uint4* ap = (const uint4*)(Ab + (wm + mt * 16 + lm) * 32);
    a[mt].q[0] = ap[hi];
    a[mt].q[1] = ap[2 + hi];
  }
#pragma unroll
  for (int nt = 0; nt < 4; nt++) {
    // B 32x16 layout: lane n=lm; chunks K[klo..klo+8), K[16+klo..+8), klo=hi*8
    const uint4* bp = (const uint4*)(Bb + (wn + nt * 16 + lm) * 32);
    b[nt].q[0] = bp[hi];
    b[nt].q[1] = bp[2 + hi];
  }
#pragma unroll
  for (int mt = 0; mt < 2; mt++)
#pragma unroll
    for (int nt = 0; nt < 4; nt++)
      acc[mt][nt] = __builtin_amdgcn_wmma_f32_16x16x32_bf16(
          false, a[mt].v, false, b[nt].v, (short)0, acc[mt][nt], false, false);
}

// C(128x128 f32) = A(rowBase..+128, K) * Bt(colBase..+128, K)^T
// A: [M][K] bf16 row-major, Bt: [N][K] bf16 (weights pre-transposed).
// 256 threads = 8 waves; wave (wm,wn) owns a 32x64 subtile = 2x4 WMMA tiles.
// Double-buffered LDS, software pipeline: barrier -> issue k+1 global loads ->
// WMMAs on tile k -> (s_wait_loadcnt) ds_store into other buffer. Last K-step
// peeled so the steady-state loop has no conditionals.
__device__ __forceinline__ void gemm_mainloop_128x128(
    const __bf16* __restrict__ A, const __bf16* __restrict__ Bt, int K,
    int rowBase, int colBase, __bf16* As, __bf16* Bs, v8f acc[2][4]) {
  const int tid  = threadIdx.x;
  const int lane = tid & 31;
  const int wave = tid >> 5;
  const int wm = (wave & 3) * 32;   // 4 waves along M
  const int wn = (wave >> 2) * 64;  // 2 waves along N
  const int lm = lane & 15;
  const int hi = lane >> 4;         // half-wave select

  // Straight-line staging: each thread owns rows (r0, r0+64), 8-elem chunk c0.
  const int r0 = tid >> 2, c0 = (tid & 3) * 8;
  const __bf16* gA = A + (size_t)(rowBase + r0) * K + c0;
  const __bf16* gB = Bt + (size_t)(colBase + r0) * K + c0;
  const size_t rowStep = (size_t)64 * K;
  const int sOff = r0 * 32 + c0;

  // Prologue: stage k=0 into buffer 0.
  *(uint4*)(As + sOff)           = *(const uint4*)(gA);
  *(uint4*)(As + sOff + 64 * 32) = *(const uint4*)(gA + rowStep);
  *(uint4*)(Bs + sOff)           = *(const uint4*)(gB);
  *(uint4*)(Bs + sOff + 64 * 32) = *(const uint4*)(gB + rowStep);

  int buf = 0;
  for (int k0 = 0; k0 < K - 32; k0 += 32) {
    __syncthreads();  // buf[buf] staged & visible; readers of buf[buf^1] done
    // Pipelined global loads for tile k0+32; consumed only after the WMMAs.
    const uint4 ra0 = *(const uint4*)(gA + k0 + 32);
    const uint4 ra1 = *(const uint4*)(gA + k0 + 32 + rowStep);
    const uint4 rb0 = *(const uint4*)(gB + k0 + 32);
    const uint4 rb1 = *(const uint4*)(gB + k0 + 32 + rowStep);
    // Pull tile-after-next toward the near caches (speculative: end overrun
    // stays inside the workspace / is dropped per ISA 10.5).
    __builtin_prefetch(gA + k0 + 64, 0, 3);
    __builtin_prefetch(gB + k0 + 64, 0, 3);

    gemm_compute_step(As + buf * 4096, Bs + buf * 4096, wm, wn, lm, hi, acc);

    __bf16* dA = As + (buf ^ 1) * 4096 + sOff;
    __bf16* dB = Bs + (buf ^ 1) * 4096 + sOff;
    *(uint4*)(dA)           = ra0;
    *(uint4*)(dA + 64 * 32) = ra1;
    *(uint4*)(dB)           = rb0;
    *(uint4*)(dB + 64 * 32) = rb1;
    buf ^= 1;
  }
  __syncthreads();  // final tile staged
  gemm_compute_step(As + buf * 4096, Bs + buf * 4096, wm, wn, lm, hi, acc);
}

// ---------------------------------------------------------------- QKV GEMM
// qkv[4096 x 6144] = x_bf[4096 x 2048] @ W_attn^T(pre-transposed) + b_attn
// Epilogue scatters: Q,K -> [bh][s][d] bf16 ; V -> transposed [bh][d][s] bf16
__global__ __launch_bounds__(256) void qkv_gemm_kernel(
    const __bf16* __restrict__ xb, const __bf16* __restrict__ wt,
    const float* __restrict__ bias, __bf16* __restrict__ qb,
    __bf16* __restrict__ kb, __bf16* __restrict__ vt) {
  __shared__ __align__(16) __bf16 As[2 * 128 * 32];
  __shared__ __align__(16) __bf16 Bs[2 * 128 * 32];
  v8f acc[2][4];
#pragma unroll
  for (int i = 0; i < 2; i++)
#pragma unroll
    for (int j = 0; j < 4; j++)
#pragma unroll
      for (int e = 0; e < 8; e++) acc[i][j][e] = 0.0f;

  const int rowBase = blockIdx.y * 128;
  const int colBase = blockIdx.x * 128;
  gemm_mainloop_128x128(xb, wt, 2048, rowBase, colBase, As, Bs, acc);

  const int lane = threadIdx.x & 31, wave = threadIdx.x >> 5;
  const int wm = (wave & 3) * 32, wn = (wave >> 2) * 64;
  const int lm = lane & 15, hi8 = (lane >> 4) * 8;
  const int sec = colBase >> 11;  // 0=Q, 1=K, 2=V (128-col blocks never straddle)

#pragma unroll
  for (int nt = 0; nt < 4; nt++) {
    const int cn = wn + nt * 16 + lm;
    const float bv = bias[colBase + cn];
    const int e = (colBase & 2047) + cn;  // channel within section
    const int h = e >> 7, d = e & 127;
#pragma unroll
    for (int mt = 0; mt < 2; mt++) {
#pragma unroll
      for (int r = 0; r < 8; r++) {
        const int gm = rowBase + wm + mt * 16 + hi8 + r;  // global row = b*S+s
        const int bb = gm >> 11, s = gm & 2047;
        const int bh = bb * 16 + h;
        const __bf16 o = f2bf(acc[mt][nt][r] + bv);
        if (sec == 0)      qb[(size_t)(bh * 2048 + s) * 128 + d] = o;
        else if (sec == 1) kb[(size_t)(bh * 2048 + s) * 128 + d] = o;
        else               vt[(size_t)(bh * 128 + d) * 2048 + s] = o;
      }
    }
  }
}

// ------------------------------------------------- flash attention (causal)
// grid (S/128, B*H). 8 independent waves/block; wave owns 16 queries,
// online softmax over key blocks of 32, O accumulated 16x128 in f32.
__global__ __launch_bounds__(256) void attn_kernel(
    const __bf16* __restrict__ qb, const __bf16* __restrict__ kb,
    const __bf16* __restrict__ vt, __bf16* __restrict__ ao) {
  __shared__ __align__(16) __bf16 Ps[8 * 16 * 32];  // per-wave P staging

  const int lane = threadIdx.x & 31, wave = threadIdx.x >> 5;
  const int lm = lane & 15, hi = lane >> 4, hi8 = hi * 8;
  const int bh = blockIdx.y;
  const int qBase = blockIdx.x * 128 + wave * 16;

  const __bf16* qrow = qb + (size_t)bh * 2048 * 128;
  const __bf16* krow = kb + (size_t)bh * 2048 * 128;
  const __bf16* vrow = vt + (size_t)bh * 128 * 2048;
  __bf16* pw = Ps + wave * 512;

  // Q fragments: 4 chunks of D=128 (16x32 each), kept in registers
  Frag qf[4];
  {
    const uint4* qp = (const uint4*)(qrow + (size_t)(qBase + lm) * 128);
#pragma unroll
    for (int kc = 0; kc < 4; kc++) {
      qf[kc].q[0] = qp[kc * 4 + hi];
      qf[kc].q[1] = qp[kc * 4 + 2 + hi];
    }
  }

  v8f oacc[8];
  float mrow[8], lrow[8];
#pragma unroll
  for (int ct = 0; ct < 8; ct++)
#pragma unroll
    for (int e = 0; e < 8; e++) oacc[ct][e] = 0.0f;
#pragma unroll
  for (int r = 0; r < 8; r++) { mrow[r] = -3.0e38f; lrow[r] = 0.0f; }

  const float scale = 0.08838834764831845f;  // 1/sqrt(128)
  const int nkb = (qBase + 16 + 31) >> 5;    // causal: keys < qBase+16

  for (int kbi = 0; kbi < nkb; kbi++) {
    const int j0 = kbi * 32;
    // S = Q @ K^T  (two 16x16 key tiles)
    v8f sacc[2];
#pragma unroll
    for (int nt = 0; nt < 2; nt++) {
#pragma unroll
      for (int e = 0; e < 8; e++) sacc[nt][e] = 0.0f;
      const uint4* kp = (const uint4*)(krow + (size_t)(j0 + nt * 16 + lm) * 128);
#pragma unroll
      for (int kc = 0; kc < 4; kc++) {
        Frag bfg;
        bfg.q[0] = kp[kc * 4 + hi];
        bfg.q[1] = kp[kc * 4 + 2 + hi];
        sacc[nt] = __builtin_amdgcn_wmma_f32_16x16x32_bf16(
            false, qf[kc].v, false, bfg.v, (short)0, sacc[nt], false, false);
      }
    }
    // online softmax per query row (rows live across 16-lane halves)
    float p[2][8], fexp[8];
#pragma unroll
    for (int r = 0; r < 8; r++) {
      const int qr = qBase + hi8 + r;
      float v0 = sacc[0][r] * scale;
      float v1 = sacc[1][r] * scale;
      if (j0 + lm > qr)      v0 = -3.0e38f;
      if (j0 + 16 + lm > qr) v1 = -3.0e38f;
      float vm = fmaxf(v0, v1);
#pragma unroll
      for (int m = 1; m < 16; m <<= 1) vm = fmaxf(vm, __shfl_xor(vm, m, 32));
      const float mnew = fmaxf(mrow[r], vm);
      const float f = __expf(mrow[r] - mnew);
      mrow[r] = mnew;
      const float p0 = __expf(v0 - mnew), p1 = __expf(v1 - mnew);
      float rs = p0 + p1;
#pragma unroll
      for (int m = 1; m < 16; m <<= 1) rs += __shfl_xor(rs, m, 32);
      lrow[r] = lrow[r] * f + rs;
      p[0][r] = p0; p[1][r] = p1; fexp[r] = f;
    }
#pragma unroll
    for (int ct = 0; ct < 8; ct++)
#pragma unroll
      for (int r = 0; r < 8; r++) oacc[ct][r] *= fexp[r];

    // P: C-layout -> A-layout through private LDS slice (same-wave, in-order)
#pragma unroll
    for (int nt = 0; nt < 2; nt++)
#pragma unroll
      for (int r = 0; r < 8; r++)
        pw[(hi8 + r) * 32 + nt * 16 + lm] = f2bf(p[nt][r]);
    Frag pf;
    {
      const uint4* pp = (const uint4*)(pw + lm * 32);
      pf.q[0] = pp[hi];
      pf.q[1] = pp[2 + hi];
    }
    // O += P @ V  (V pre-transposed: contiguous keys per d-row)
#pragma unroll
    for (int ct = 0; ct < 8; ct++) {
      const uint4* vp =
          (const uint4*)(vrow + (size_t)(ct * 16 + lm) * 2048 + j0);
      Frag vf;
      vf.q[0] = vp[hi];
      vf.q[1] = vp[2 + hi];
      oacc[ct] = __builtin_amdgcn_wmma_f32_16x16x32_bf16(
          false, pf.v, false, vf.v, (short)0, oacc[ct], false, false);
    }
  }

  // finalize: O /= l, store bf16 into attn_out [b*S+s][h*128+d]
  const int bb = bh >> 4, h = bh & 15;
#pragma unroll
  for (int r = 0; r < 8; r++) {
    const float inv = 1.0f / lrow[r];
    const int qr = qBase + hi8 + r;
    __bf16* orow = ao + (size_t)(bb * 2048 + qr) * 2048 + h * 128;
#pragma unroll
    for (int ct = 0; ct < 8; ct++) orow[ct * 16 + lm] = f2bf(oacc[ct][r] * inv);
  }
}

// ------------------------------------------------------------- output proj
__global__ __launch_bounds__(256) void proj_gemm_kernel(
    const __bf16* __restrict__ ab, const __bf16* __restrict__ wt,
    const float* __restrict__ bias, float* __restrict__ out) {
  __shared__ __align__(16) __bf16 As[2 * 128 * 32];
  __shared__ __align__(16) __bf16 Bs[2 * 128 * 32];
  v8f acc[2][4];
#pragma unroll
  for (int i = 0; i < 2; i++)
#pragma unroll
    for (int j = 0; j < 4; j++)
#pragma unroll
      for (int e = 0; e < 8; e++) acc[i][j][e] = 0.0f;

  const int rowBase = blockIdx.y * 128;
  const int colBase = blockIdx.x * 128;
  gemm_mainloop_128x128(ab, wt, 2048, rowBase, colBase, As, Bs, acc);

  const int lane = threadIdx.x & 31, wave = threadIdx.x >> 5;
  const int wm = (wave & 3) * 32, wn = (wave >> 2) * 64;
  const int lm = lane & 15, hi8 = (lane >> 4) * 8;

#pragma unroll
  for (int nt = 0; nt < 4; nt++) {
    const int gn = colBase + wn + nt * 16 + lm;
    const float bv = bias[gn];
#pragma unroll
    for (int mt = 0; mt < 2; mt++) {
#pragma unroll
      for (int r = 0; r < 8; r++) {
        const int gm = rowBase + wm + mt * 16 + hi8 + r;
        out[(size_t)gm * 2048 + gn] = acc[mt][nt][r] + bv;
      }
    }
  }
}

// ------------------------------------------------------------------ launch
extern "C" void kernel_launch(void* const* d_in, const int* in_sizes, int n_in,
                              void* d_out, int out_size, void* d_ws,
                              size_t ws_size, hipStream_t stream) {
  const float* x      = (const float*)d_in[0];  // (2,2048,2048)
  const float* W_attn = (const float*)d_in[1];  // (2048,6144)
  const float* b_attn = (const float*)d_in[2];  // (6144,)
  const float* W_proj = (const float*)d_in[3];  // (2048,2048)
  const float* b_proj = (const float*)d_in[4];  // (2048,)
  float* out = (float*)d_out;

  char* ws = (char*)d_ws;
  __bf16* xb  = (__bf16*)(ws);                        // 16 MB  x bf16
  __bf16* wat = (__bf16*)(ws + (16ll << 20));         // 24 MB  W_attn^T bf16
  __bf16* wpt = (__bf16*)(ws + (40ll << 20));         //  8 MB  W_proj^T bf16
  __bf16* qb  = (__bf16*)(ws + (48ll << 20));         // 16 MB  Q [bh][s][d]
  __bf16* kbuf= (__bf16*)(ws + (64ll << 20));         // 16 MB  K [bh][s][d]
  __bf16* vt  = (__bf16*)(ws + (80ll << 20));         // 16 MB  V^T [bh][d][s]
  __bf16* ao  = (__bf16*)(ws + (96ll << 20));         // 16 MB  attn out bf16

  const int NX = 4096 * 2048;
  cvt_bf16_kernel<<<NX / (8 * 256), 256, 0, stream>>>(x, xb, NX);
  transpose_cvt_kernel<<<dim3(6144 / 32, 2048 / 32), dim3(32, 8), 0, stream>>>(
      W_attn, wat, 2048, 6144);
  transpose_cvt_kernel<<<dim3(2048 / 32, 2048 / 32), dim3(32, 8), 0, stream>>>(
      W_proj, wpt, 2048, 2048);

  qkv_gemm_kernel<<<dim3(48, 32), 256, 0, stream>>>(xb, wat, b_attn, qb, kbuf, vt);
  attn_kernel<<<dim3(16, 32), 256, 0, stream>>>(qb, kbuf, vt, ao);
  proj_gemm_kernel<<<dim3(16, 32), 256, 0, stream>>>(ao, wpt, b_proj, out);
}
